// SkipAttention_34763465294430
// MI455X (gfx1250) — matrix operand order repaired and checked by reading, not compiled, not run on hardware
//
#include <hip/hip_runtime.h>
#include <hip/hip_bf16.h>
#include <math.h>

typedef __attribute__((ext_vector_type(16))) _Float16 v16h;
typedef __attribute__((ext_vector_type(8)))  _Float16 v8h;
typedef __attribute__((ext_vector_type(8)))  float    v8f;
typedef __attribute__((ext_vector_type(4)))  unsigned int u32x4;
typedef __attribute__((ext_vector_type(8)))  int          i32x8;
typedef __attribute__((ext_vector_type(4)))  int          i32x4;

#define HDIM 512
#define TSEQ 2048
#define NBATCH 8

// ---------------------------------------------------------------------------
// WMMA helper: D = A(16x32 f16) * B(32x16 f16) + C(16x16 f32)
// ---------------------------------------------------------------------------
__device__ __forceinline__ v8f wmma_f16(v16h a, v16h b, v8f c) {
    return __builtin_amdgcn_wmma_f32_16x16x32_f16(
        /*neg_a=*/false, a, /*neg_b=*/false, b,
        /*c_mod=*/(short)0, c, /*reuse_a=*/false, /*reuse_b=*/false);
}

// A-matrix fragment (16x32, f16), row-major source with leading dim `ld`.
// ISA layout: lanes 0-15 hold rows M=0..15 with K = 0..7 & 16..23,
//             lanes 16-31 hold the same rows with K = 8..15 & 24..31.
__device__ __forceinline__ v16h load_fragA(const _Float16* p, int row0, int ld, int k0) {
    int lane = threadIdx.x & 31;
    int m = lane & 15;
    int g = lane >> 4;                       // 0 or 1
    const _Float16* base = p + (size_t)(row0 + m) * ld + k0 + g * 8;
    v8h lo = *(const v8h*)(base);            // K = k0+g*8   .. +7
    v8h hi = *(const v8h*)(base + 16);       // K = k0+g*8+16.. +7
    v16h r;
#pragma unroll
    for (int i = 0; i < 8; ++i) { r[i] = lo[i]; r[i + 8] = hi[i]; }
    return r;
}

// B-matrix fragment (32x16, f16) from a K-major source: element [k][n] is
// read from p[(row0+n)*ld + k0 + k]. Layout matches A fragment:
// lanes 0-15 hold K=0..7 & 16..23, lanes 16-31 hold K=8..15 & 24..31, column N = lane%16.
__device__ __forceinline__ v16h load_fragB(const _Float16* p, int row0, int ld, int k0) {
    int lane = threadIdx.x & 31;
    int n = lane & 15;
    int g = lane >> 4;
    const _Float16* base = p + (size_t)(row0 + n) * ld + k0 + g * 8;
    v8h lo = *(const v8h*)(base);
    v8h hi = *(const v8h*)(base + 16);
    v16h r;
#pragma unroll
    for (int i = 0; i < 8; ++i) { r[i] = lo[i]; r[i + 8] = hi[i]; }
    return r;
}

// ---------------------------------------------------------------------------
// TDM: DMA a rows x cols (f16) tile from global into LDS (per-wave).
// Builds the D# descriptor in SGPRs (readfirstlane-forced uniform).
// ---------------------------------------------------------------------------
__device__ __forceinline__ void tdm_load_tile_f16(const _Float16* gsrc, void* lds_dst,
                                                  int cols, int rows,
                                                  int tensor_rows, int row_stride) {
    unsigned long long ga = (unsigned long long)(size_t)(const void*)gsrc;
    unsigned lds_off = (unsigned)(size_t)lds_dst;   // LDS aperture: low 32 bits

    u32x4 g0;
    g0[0] = (unsigned)__builtin_amdgcn_readfirstlane(1);                 // count=1, no gather
    g0[1] = (unsigned)__builtin_amdgcn_readfirstlane((int)lds_off);      // lds_addr
    g0[2] = (unsigned)__builtin_amdgcn_readfirstlane((int)(unsigned)ga); // global_addr[31:0]
    g0[3] = (unsigned)__builtin_amdgcn_readfirstlane(
                (int)((unsigned)(ga >> 32) | (2u << 30)));               // ga[56:32] | type=2

    i32x8 g1;
    g1[0] = __builtin_amdgcn_readfirstlane(0x10000);               // data_size=1 (2B), mask=0
    g1[1] = __builtin_amdgcn_readfirstlane((int)((unsigned)cols << 16));        // tensor_dim0 lo16
    g1[2] = __builtin_amdgcn_readfirstlane((int)((unsigned)tensor_rows << 16)); // tensor_dim1 lo16
    g1[3] = __builtin_amdgcn_readfirstlane((int)((unsigned)cols << 16));        // tile_dim0
    g1[4] = __builtin_amdgcn_readfirstlane(rows);                  // tile_dim1, tile_dim2=0
    g1[5] = __builtin_amdgcn_readfirstlane(row_stride);            // tensor_dim0_stride lo32
    g1[6] = __builtin_amdgcn_readfirstlane(0);
    g1[7] = __builtin_amdgcn_readfirstlane(0);

    i32x4 z4 = {0, 0, 0, 0};
#if __clang_major__ >= 23
    i32x8 z8 = {0, 0, 0, 0, 0, 0, 0, 0};
    __builtin_amdgcn_tensor_load_to_lds(g0, g1, z4, z4, z8, 0);
#else
    __builtin_amdgcn_tensor_load_to_lds(g0, g1, z4, z4, 0);
#endif
}

// ---------------------------------------------------------------------------
// Stage 1: f32 -> f16 conversion
// ---------------------------------------------------------------------------
__global__ void cvt_f32_to_f16(const float* __restrict__ src,
                               _Float16* __restrict__ dst, int n) {
    int i = blockIdx.x * blockDim.x + threadIdx.x;
    if (i < n) dst[i] = (_Float16)src[i];
}

// ---------------------------------------------------------------------------
// Stage 2: projection GEMM  y[m][n] = sum_k xh[m][k] * W[n][k] + b[n]
// mode 0: Q (scaled by 1/sqrt(H)), row-major f16 out
// mode 1: K, row-major f16 out
// mode 2: V, transposed out Vt[b][d][t] (f16)
// One wave per 16x16 output tile; 4 waves / 128-thread block.
// ---------------------------------------------------------------------------
__global__ void __launch_bounds__(128)
proj_gemm(const _Float16* __restrict__ xh, const _Float16* __restrict__ wh,
          const float* __restrict__ bias, _Float16* __restrict__ out,
          int mode, float scale) {
    int wave = threadIdx.x >> 5;
    int lane = threadIdx.x & 31;
    int tile = blockIdx.x * 4 + wave;        // 0 .. 32767
    int tn = tile & 31;                      // 512/16 = 32 N tiles
    int tm = tile >> 5;                      // 16384/16 = 1024 M tiles
    int n0 = tn << 4, m0 = tm << 4;

    v8f c = {};
#pragma unroll
    for (int k = 0; k < 16; ++k) {
        v16h a = load_fragA(xh, m0, HDIM, k * 32);
        v16h b = load_fragB(wh, n0, HDIM, k * 32);
        c = wmma_f16(a, b, c);
    }

    int col = lane & 15;
    int g   = lane >> 4;
    float bb = bias[n0 + col];

    if (mode == 2) {
        // transposed store: Vt[batch][d][t]; per lane v=0..7 is contiguous in t
        int tloc = m0 & (TSEQ - 1);
        int bidx = m0 >> 11;
        v8h st;
#pragma unroll
        for (int v = 0; v < 8; ++v) st[v] = (_Float16)(c[v] + bb);
        size_t idx = ((size_t)bidx * HDIM + n0 + col) * TSEQ + tloc + g * 8;
        *(v8h*)(out + idx) = st;
    } else {
#pragma unroll
        for (int v = 0; v < 8; ++v) {
            int r = v + g * 8;
            out[(size_t)(m0 + r) * HDIM + n0 + col] = (_Float16)((c[v] + bb) * scale);
        }
    }
}

// ---------------------------------------------------------------------------
// Stage 3: flash attention + residual.
// Each wave owns 16 query rows; Q tile staged into LDS via the Tensor Data
// Mover; full 16x512 f32 output accumulator held in VGPRs (32 v8f tiles);
// K/V streamed in s-tiles of 32 with online softmax. 2 waves / 64-th block.
// ---------------------------------------------------------------------------
__global__ void __launch_bounds__(64)
flash_attn(const _Float16* __restrict__ Qh, const _Float16* __restrict__ Kh,
           const _Float16* __restrict__ Vt, const float* __restrict__ xres,
           float* __restrict__ out) {
    __shared__ _Float16 qs[2][16 * HDIM];    // 2 x 16KB Q tiles
    __shared__ _Float16 ps[2][16 * 32];      // 2 x 1KB  P tiles

    int wave = threadIdx.x >> 5;
    int lane = threadIdx.x & 31;
    int tile = blockIdx.x * 2 + wave;        // 0 .. 1023 row tiles
    int bidx = tile >> 7;                    // 128 tiles per batch
    int t0   = (tile & 127) << 4;

    const _Float16* Qb = Qh + (size_t)bidx * TSEQ * HDIM;
    const _Float16* Kb = Kh + (size_t)bidx * TSEQ * HDIM;
    const _Float16* Vb = Vt + (size_t)bidx * HDIM * TSEQ;

    _Float16* q = qs[wave];
    _Float16* p = ps[wave];

    // Stage Q tile (16 rows x 512 cols, f16) into LDS via TDM.
    tdm_load_tile_f16(Qb + (size_t)t0 * HDIM, (void*)q,
                      /*cols=*/HDIM, /*rows=*/16,
                      /*tensor_rows=*/TSEQ, /*row_stride=*/HDIM);
    __builtin_amdgcn_s_wait_tensorcnt(0);
    __syncthreads();

    v8f zero = {};
    v8f acc[32];
#pragma unroll
    for (int i = 0; i < 32; ++i) acc[i] = zero;
    float mrow[8], lrow[8];
#pragma unroll
    for (int v = 0; v < 8; ++v) { mrow[v] = -3.0e38f; lrow[v] = 0.0f; }

    int col = lane & 15;
    int g   = lane >> 4;

    for (int s0 = 0; s0 < TSEQ; s0 += 32) {
        if (s0 + 32 < TSEQ) {  // prefetch next K tile rows (global_prefetch_b8)
            __builtin_prefetch(Kb + (size_t)(s0 + 32 + col) * HDIM, 0, 1);
            __builtin_prefetch(Kb + (size_t)(s0 + 48 + col) * HDIM, 0, 1);
        }
        // --- scores: S = Q * K^T for 16 rows x 32 cols, f32 accum ---
        v8f sc0 = zero, sc1 = zero;
#pragma unroll
        for (int kd = 0; kd < 16; ++kd) {
            v16h a  = load_fragA(q, 0, HDIM, kd * 32);
            v16h b0 = load_fragB(Kb, s0,      HDIM, kd * 32);
            v16h b1 = load_fragB(Kb, s0 + 16, HDIM, kd * 32);
            sc0 = wmma_f16(a, b0, sc0);
            sc1 = wmma_f16(a, b1, sc1);
        }
        // --- online softmax (scale already folded into Q) ---
#pragma unroll
        for (int v = 0; v < 8; ++v) {
            float mv = fmaxf(sc0[v], sc1[v]);
#pragma unroll
            for (int off = 8; off >= 1; off >>= 1)
                mv = fmaxf(mv, __shfl_xor(mv, off, 32));
            float mn = fmaxf(mrow[v], mv);
            float alpha = __expf(mrow[v] - mn);
            float p0 = __expf(sc0[v] - mn);
            float p1 = __expf(sc1[v] - mn);
            float rs = p0 + p1;
#pragma unroll
            for (int off = 8; off >= 1; off >>= 1)
                rs += __shfl_xor(rs, off, 32);
            lrow[v] = lrow[v] * alpha + rs;
            mrow[v] = mn;
#pragma unroll
            for (int dt = 0; dt < 32; ++dt) acc[dt][v] *= alpha;
            int row = v + g * 8;
            p[row * 32 + col]      = (_Float16)p0;
            p[row * 32 + col + 16] = (_Float16)p1;
        }
        __syncthreads();   // P tile visible in A-fragment order
        // --- out += P(16x32) * V(32x512) ---
        v16h pa = load_fragA(p, 0, 32, 0);
#pragma unroll
        for (int dt = 0; dt < 32; ++dt) {
            v16h vb = load_fragB(Vb, dt * 16, TSEQ, s0);
            acc[dt] = wmma_f16(pa, vb, acc[dt]);
        }
        __syncthreads();   // protect P tile WAR before next iteration
    }

    // --- epilogue: normalize + residual ---
#pragma unroll
    for (int dt = 0; dt < 32; ++dt) {
#pragma unroll
        for (int v = 0; v < 8; ++v) {
            int r = v + g * 8;
            int d = dt * 16 + col;
            size_t idx = ((size_t)bidx * TSEQ + t0 + r) * HDIM + d;
            out[idx] = acc[dt][v] / lrow[v] + xres[idx];
        }
    }
}

// ---------------------------------------------------------------------------
// Launch
// ---------------------------------------------------------------------------
extern "C" void kernel_launch(void* const* d_in, const int* in_sizes, int n_in,
                              void* d_out, int out_size, void* d_ws, size_t ws_size,
                              hipStream_t stream) {
    (void)in_sizes; (void)n_in; (void)out_size; (void)ws_size;
    const float* x  = (const float*)d_in[0];
    const float* Wq = (const float*)d_in[1];
    const float* bq = (const float*)d_in[2];
    const float* Wk = (const float*)d_in[3];
    const float* bk = (const float*)d_in[4];
    const float* Wv = (const float*)d_in[5];
    const float* bv = (const float*)d_in[6];
    float* out = (float*)d_out;

    const size_t NX = (size_t)NBATCH * TSEQ * HDIM;  // 8,388,608
    const size_t NW = (size_t)HDIM * HDIM;           // 262,144

    _Float16* ws  = (_Float16*)d_ws;
    _Float16* xh  = ws;
    _Float16* wqh = xh  + NX;
    _Float16* wkh = wqh + NW;
    _Float16* wvh = wkh + NW;
    _Float16* Qh  = wvh + NW;
    _Float16* Kh  = Qh  + NX;
    _Float16* Vt  = Kh  + NX;

    cvt_f32_to_f16<<<(int)((NX + 255) / 256), 256, 0, stream>>>(x,  xh,  (int)NX);
    cvt_f32_to_f16<<<(int)((NW + 255) / 256), 256, 0, stream>>>(Wq, wqh, (int)NW);
    cvt_f32_to_f16<<<(int)((NW + 255) / 256), 256, 0, stream>>>(Wk, wkh, (int)NW);
    cvt_f32_to_f16<<<(int)((NW + 255) / 256), 256, 0, stream>>>(Wv, wvh, (int)NW);

    const float scale = 0.04419417382415922f;  // 1/sqrt(512)
    proj_gemm<<<8192, 128, 0, stream>>>(xh, wqh, bq, Qh, 0, scale);
    proj_gemm<<<8192, 128, 0, stream>>>(xh, wkh, bk, Kh, 1, 1.0f);
    proj_gemm<<<8192, 128, 0, stream>>>(xh, wvh, bv, Vt, 2, 1.0f);

    flash_attn<<<512, 64, 0, stream>>>(Qh, Kh, Vt, x, out);
}
